// ResidualVectorQuantizer_33535104647393
// MI455X (gfx1250) — matrix operand order repaired and checked by reading, not compiled, not run on hardware
//
#include <hip/hip_runtime.h>
#include <cstdint>
#include <cstddef>

// ---------------------------------------------------------------------------
// Residual Vector Quantizer, fused 4-stage kernel for gfx1250 (MI455X).
// Distances via bf16 WMMA (v_wmma_f32_16x16x32_bf16), fp32 norms/update.
// ---------------------------------------------------------------------------

typedef __attribute__((ext_vector_type(16))) __bf16        v16bf;
typedef __attribute__((ext_vector_type(8)))  float         v8f;
typedef __attribute__((ext_vector_type(8)))  unsigned int  v8u;

#define RVQ_STAGES   4
#define RVQ_NUM_EMB  1024
#define RVQ_DIM      128
#define RVQ_B        16
#define RVQ_T        4096
#define ROWS_PER_WAVE  16
#define WAVES_PER_BLK  8
#define ROWS_PER_BLK   (ROWS_PER_WAVE * WAVES_PER_BLK)   // 128
#define CHUNK_CODES    128
#define ROW_WORDS      68   // 64 dwords of bf16 row + 4 pad (bank-conflict break)

// round-to-nearest-even f32 -> bf16, packed pair into one dword
__device__ __forceinline__ unsigned int pack_bf16(float a, float b) {
    unsigned int ua = __float_as_uint(a);
    unsigned int ub = __float_as_uint(b);
    ua += 0x7FFFu + ((ua >> 16) & 1u);
    ub += 0x7FFFu + ((ub >> 16) & 1u);
    return (ua >> 16) | (ub & 0xFFFF0000u);
}

__global__ void rvq_init_loss(float* loss) { *loss = 0.0f; }

__global__ __launch_bounds__(256) void rvq_fused_kernel(
    const float* __restrict__ z_e,        // [16,128,4096]
    const float* __restrict__ codebooks,  // [4,1024,128]
    float*       __restrict__ out_q,      // [16,128,4096]
    float*       __restrict__ out_loss,   // [1]
    int*         __restrict__ out_idx)    // [16,4,4096]
{
    __shared__ unsigned int cbLDS[CHUNK_CODES * ROW_WORDS]; // bf16 codes, padded rows
    __shared__ float        cnorm[CHUNK_CODES];
    __shared__ int          idxScr[WAVES_PER_BLK][16];

    const int tid  = threadIdx.x;
    const int w    = tid >> 5;        // wave in block
    const int lane = tid & 31;
    const int half = lane >> 4;       // 0: lanes 0-15, 1: lanes 16-31
    const int lm   = lane & 15;

    const int b  = blockIdx.x >> 5;                  // batch
    const int t0 = (blockIdx.x & 31) * ROWS_PER_BLK; // first t of block
    const int trow = t0 + w * ROWS_PER_WAVE + lm;    // this lane's row (M = lm)
    const size_t zbase = (size_t)b * RVQ_DIM * RVQ_T;

    // ---- load residual = z_e rows, in WMMA A-fragment register layout ----
    // element (c,e): k = 32c + (e&7) + 16*(e>>3) + 8*half   (lane row M = lm)
    float r[4][16];
#pragma unroll
    for (int c = 0; c < 4; ++c) {
#pragma unroll
        for (int e = 0; e < 16; ++e) {
            const int k = 32 * c + (e & 7) + 16 * (e >> 3) + 8 * half;
            r[c][e] = z_e[zbase + (size_t)k * RVQ_T + trow];
        }
    }

    float lossLocal = 0.0f;

    for (int s = 0; s < RVQ_STAGES; ++s) {
        const float* cbS = codebooks + (size_t)s * RVQ_NUM_EMB * RVQ_DIM;

        // ---- build bf16 A-fragments from fp32 residual ----
        v16bf afrag[4];
#pragma unroll
        for (int c = 0; c < 4; ++c) {
            v8u u;
#pragma unroll
            for (int wd = 0; wd < 8; ++wd)
                u[wd] = pack_bf16(r[c][2 * wd], r[c][2 * wd + 1]);
            afrag[c] = __builtin_bit_cast(v16bf, u);
        }

        float best[8];
        int   bidx[8];
#pragma unroll
        for (int j = 0; j < 8; ++j) { best[j] = 3.4e38f; bidx[j] = 0; }

        for (int h = 0; h < RVQ_NUM_EMB / CHUNK_CODES; ++h) {
            __syncthreads();                       // previous chunk fully consumed
            if (tid < CHUNK_CODES) cnorm[tid] = 0.0f;
            __syncthreads();

            // ---- cooperative load of 128 codes: fp32 -> bf16 LDS + norms ----
            // flat float4 index: whole wave shares one code row -> coalesced 512B
#pragma unroll
            for (int i = 0; i < 16; ++i) {
                const int flat = tid + 256 * i;    // 0..4095
                const int cl   = flat >> 5;        // local code 0..127
                const int d4   = flat & 31;        // float4 within row
                const float4 v =
                    ((const float4*)(cbS + (size_t)(h * CHUNK_CODES + cl) * RVQ_DIM))[d4];
                cbLDS[cl * ROW_WORDS + 2 * d4]     = pack_bf16(v.x, v.y);
                cbLDS[cl * ROW_WORDS + 2 * d4 + 1] = pack_bf16(v.z, v.w);
                float ns = v.x * v.x + v.y * v.y + v.z * v.z + v.w * v.w;
#pragma unroll
                for (int off = 16; off > 0; off >>= 1)
                    ns += __shfl_xor(ns, off, 32);
                if (lane == 0) atomicAdd(&cnorm[cl], ns);
            }
            __syncthreads();

            // ---- distance tiles: 8 tiles x 16 codes, K=128 via 4 chained WMMAs ----
#pragma unroll
            for (int tile = 0; tile < CHUNK_CODES / 16; ++tile) {
                const int codeLocal = tile * 16 + lm;          // B column N = lm
                const unsigned int* rowp = &cbLDS[codeLocal * ROW_WORDS + 4 * half];
                v8f acc = {};
#pragma unroll
                for (int c = 0; c < 4; ++c) {
                    const uint4 q0 = *(const uint4*)(rowp + 16 * c);
                    const uint4 q1 = *(const uint4*)(rowp + 16 * c + 8);
                    v8u bu;
                    bu[0] = q0.x; bu[1] = q0.y; bu[2] = q0.z; bu[3] = q0.w;
                    bu[4] = q1.x; bu[5] = q1.y; bu[6] = q1.z; bu[7] = q1.w;
                    const v16bf bfrag = __builtin_bit_cast(v16bf, bu);
                    acc = __builtin_amdgcn_wmma_f32_16x16x32_bf16(
                        false, afrag[c], false, bfrag, (short)0, acc, false, false);
                }
                const float cn   = cnorm[codeLocal];
                const int   code = h * CHUNK_CODES + codeLocal;
#pragma unroll
                for (int j = 0; j < 8; ++j) {
                    const float score = cn - 2.0f * acc[j];    // ||c||^2 - 2 x.c
                    if (score < best[j]) { best[j] = score; bidx[j] = code; }
                }
            }
        }

        // ---- argmin across the 16 lanes sharing each row (per half-wave) ----
#pragma unroll
        for (int off = 1; off < 16; off <<= 1) {
#pragma unroll
            for (int j = 0; j < 8; ++j) {
                const float ob = __shfl_xor(best[j], off, 32);
                const int   oi = __shfl_xor(bidx[j], off, 32);
                if (ob < best[j]) { best[j] = ob; bidx[j] = oi; }
            }
        }
        // lower half holds rows 0..7 (slot j -> row j), upper half rows 8..15
        if (lane == 0) {
#pragma unroll
            for (int j = 0; j < 8; ++j) idxScr[w][j] = bidx[j];
        }
        if (lane == 16) {
#pragma unroll
            for (int j = 0; j < 8; ++j) idxScr[w][8 + j] = bidx[j];
        }
        __syncthreads();

        // ---- emit indices: all_indices[b][s][t] ----
        if (lane < 16)
            out_idx[((size_t)b * RVQ_STAGES + s) * RVQ_T + t0 + w * 16 + lane] =
                idxScr[w][lane];

        // ---- gather winning code row (fp32) and update residual ----
        const int rowIdx = idxScr[w][lm];
        const float* qrow = cbS + (size_t)rowIdx * RVQ_DIM;
#pragma unroll
        for (int c = 0; c < 4; ++c) {
            const int k0 = 32 * c + 8 * half;
            const float4 q0 = *(const float4*)(qrow + k0);
            const float4 q1 = *(const float4*)(qrow + k0 + 4);
            const float4 q2 = *(const float4*)(qrow + k0 + 16);
            const float4 q3 = *(const float4*)(qrow + k0 + 20);
            r[c][0]  -= q0.x; r[c][1]  -= q0.y; r[c][2]  -= q0.z; r[c][3]  -= q0.w;
            r[c][4]  -= q1.x; r[c][5]  -= q1.y; r[c][6]  -= q1.z; r[c][7]  -= q1.w;
            r[c][8]  -= q2.x; r[c][9]  -= q2.y; r[c][10] -= q2.z; r[c][11] -= q2.w;
            r[c][12] -= q3.x; r[c][13] -= q3.y; r[c][14] -= q3.z; r[c][15] -= q3.w;
        }

        // loss_s = mean(next_residual^2): accumulate local sum of squares
        float ss = 0.0f;
#pragma unroll
        for (int c = 0; c < 4; ++c)
#pragma unroll
            for (int e = 0; e < 16; ++e) ss += r[c][e] * r[c][e];
        lossLocal += ss;
    }

    // ---- loss: wave-reduce, one atomic per wave ----
    // total = (COMMIT/NUM_STAGES) * sum_s mean(r_next^2)
    {
        float ls = lossLocal;
#pragma unroll
        for (int off = 16; off > 0; off >>= 1)
            ls += __shfl_xor(ls, off, 32);
        const float scale =
            (0.25f / (float)RVQ_STAGES) / ((float)RVQ_B * RVQ_DIM * RVQ_T);
        if (lane == 0) atomicAdd(out_loss, ls * scale);
    }

    // ---- quantized_output = z_e - final residual ----
#pragma unroll
    for (int c = 0; c < 4; ++c) {
#pragma unroll
        for (int e = 0; e < 16; ++e) {
            const int k = 32 * c + (e & 7) + 16 * (e >> 3) + 8 * half;
            const size_t addr = zbase + (size_t)k * RVQ_T + trow;
            out_q[addr] = z_e[addr] - r[c][e];
        }
    }
}

extern "C" void kernel_launch(void* const* d_in, const int* in_sizes, int n_in,
                              void* d_out, int out_size, void* d_ws, size_t ws_size,
                              hipStream_t stream) {
    (void)in_sizes; (void)n_in; (void)d_ws; (void)ws_size; (void)out_size;
    const float* z_e = (const float*)d_in[0];
    const float* cbs = (const float*)d_in[1];

    float* out_q    = (float*)d_out;                              // 16*128*4096
    float* out_loss = out_q + (size_t)RVQ_B * RVQ_DIM * RVQ_T;    // [1]
    int*   out_idx  = (int*)(out_loss + 1);                       // 16*4*4096

    hipLaunchKernelGGL(rvq_init_loss, dim3(1), dim3(1), 0, stream, out_loss);

    const int nBlocks = (RVQ_B * RVQ_T) / ROWS_PER_BLK;           // 512
    hipLaunchKernelGGL(rvq_fused_kernel, dim3(nBlocks), dim3(256), 0, stream,
                       z_e, cbs, out_q, out_loss, out_idx);
}
